// WOSMnist_72842645340325
// MI455X (gfx1250) — matrix operand order repaired and checked by reading, not compiled, not run on hardware
//
#include <hip/hip_runtime.h>
#include <hip/hip_bf16.h>

typedef __attribute__((ext_vector_type(16))) _Float16 v16h;
typedef __attribute__((ext_vector_type(8)))  float    v8f;
typedef __attribute__((ext_vector_type(4)))  float    v4f;

#define SENT2 (-60000.0f)   // finite in f16: sorts below all real values, *0 weight = 0
#define SENTF (-1.0e30f)    // fc sort sentinel (stays f32, never enters WMMA)
#define L2_STRIDE 265       // odd stride -> conflict-free column access across 16 rows

// ---------------- Layer 1: WOS conv 3x3 on 1 channel (sort-9 + 9x16 dot) -------------
__global__ __launch_bounds__(256) void l1_kernel(const float* __restrict__ xin,
                                                 const float* __restrict__ w1,
                                                 float* __restrict__ h1) {
    int idx = blockIdx.x * 256 + threadIdx.x;
    if (idx >= 512 * 676) return;
    int b = idx / 676, p = idx % 676;
    int y = p / 26, x0 = p % 26;
    const float* src = xin + (b * 28 + y) * 28 + x0;
    float v[9];
#pragma unroll
    for (int dy = 0; dy < 3; ++dy)
#pragma unroll
        for (int dx = 0; dx < 3; ++dx) v[dy * 3 + dx] = src[dy * 28 + dx];
    // descending bubble network (36 CE)
#pragma unroll
    for (int i = 0; i < 8; ++i)
#pragma unroll
        for (int j = 0; j < 8 - i; ++j) {
            float a = v[j], c = v[j + 1];
            v[j] = fmaxf(a, c); v[j + 1] = fminf(a, c);
        }
#pragma unroll
    for (int o = 0; o < 16; ++o) {
        float acc = 0.f;
#pragma unroll
        for (int j = 0; j < 9; ++j) acc += v[j] * w1[o * 9 + j];
        h1[((b * 16 + o) * 26 + y) * 26 + x0] = acc;
    }
}

// ---- init: pack weights into per-lane WMMA B layout: bpk[chunk][lane][16 halfs] ----
// lane L: n = L&15, k = chunk*32 + (L>=16 ? 8 : 0) + (kk < 8 ? kk : kk + 8)  => one v16h load per lane.
__global__ __launch_bounds__(256) void initw_kernel(const float* __restrict__ w2,
                                                    const float* __restrict__ w3,
                                                    _Float16* __restrict__ w2pk,
                                                    _Float16* __restrict__ w3pk) {
    int i = blockIdx.x * 256 + threadIdx.x;
    if (i < 5 * 512) {                       // w2: K padded 144->160, 5 chunks
        int ch = i >> 9, lane = (i >> 4) & 31, kk = i & 15;
        int n = lane & 15;
        int k = ch * 32 + ((lane >= 16) ? 8 : 0) + (kk < 8 ? kk : kk + 8);
        w2pk[i] = (k < 144) ? (_Float16)w2[n * 144 + k] : (_Float16)0.0f;
    }
    if (i < 72 * 512) {                      // w3: K = 2304 = 72 chunks exactly
        int ch = i >> 9, lane = (i >> 4) & 31, kk = i & 15;
        int n = lane & 15;
        int k = ch * 32 + ((lane >= 16) ? 8 : 0) + (kk < 8 ? kk : kk + 8);
        w3pk[i] = (_Float16)w3[n * 2304 + k];
    }
}

// ---------------- Layer 2: gather 144-patch, bitonic sort rows, WMMA 16x16x(5*32) ----
__global__ __launch_bounds__(256) void l2_kernel(const float* __restrict__ h1,
                                                 const _Float16* __restrict__ w2pk,
                                                 float* __restrict__ h2) {
    __shared__ float lds[16 * L2_STRIDE];
    const int tid = threadIdx.x;
    const int tile = blockIdx.x;      // 512*36 tiles
    const int b = tile / 36;
    const int pt = tile % 36;
    { // gather: 16 threads per row, each loads one channel's 3x3 window
        const int m = tid >> 4, c = tid & 15;
        const int p = pt * 16 + m;
        const int y = p / 24, x = p % 24;
        const float* src = h1 + ((b * 16 + c) * 26 + y) * 26 + x;
        float* dst = &lds[m * L2_STRIDE + c * 9];
#pragma unroll
        for (int dy = 0; dy < 3; ++dy)
#pragma unroll
            for (int dx = 0; dx < 3; ++dx) dst[dy * 3 + dx] = src[dy * 26 + dx];
    }
    for (int i = tid; i < 16 * 112; i += 256) {     // pad 144..255 with sentinel
        int m = i / 112, e = 144 + i % 112;
        lds[m * L2_STRIDE + e] = SENT2;
    }
    __syncthreads();
    // bitonic descending sort, 16 independent rows of 256
    for (int k = 2; k <= 256; k <<= 1) {
        for (int j = k >> 1; j > 0; j >>= 1) {
            for (int q = tid; q < 16 * 128; q += 256) {
                int m = q >> 7, r = q & 127;
                int e = ((r & ~(j - 1)) << 1) | (r & (j - 1));
                int f = e | j;
                float* base = &lds[m * L2_STRIDE];
                float a = base[e], c = base[f];
                bool up = ((e & k) == 0);
                if (up ? (a < c) : (a > c)) { base[e] = c; base[f] = a; }
            }
            __syncthreads();
        }
    }
    // wave 0: D[16x16] = A[16x160] * B[160x16] via 5 chained v_wmma_f32_16x16x32_f16
    if (tid < 32) {
        const int l = tid & 15;
        const bool hi = tid >= 16;
        const v16h* bvec = (const v16h*)w2pk;
        v8f c = {};
#pragma unroll
        for (int ch = 0; ch < 5; ++ch) {
            const int kb = ch * 32;
            v16h a;
            const float* arow = &lds[l * L2_STRIDE + kb + (hi ? 8 : 0)];
#pragma unroll
            for (int kk = 0; kk < 8; ++kk) {
                a[kk]     = (_Float16)arow[kk];
                a[kk + 8] = (_Float16)arow[16 + kk];
            }
            v16h bm = bvec[ch * 32 + tid];            // 32B contiguous per lane
            c = __builtin_amdgcn_wmma_f32_16x16x32_f16(false, a, false, bm,
                                                       (short)0, c, false, false);
        }
#pragma unroll
        for (int v = 0; v < 8; ++v) {
            int m = v + (hi ? 8 : 0);
            int p = pt * 16 + m;
            int y = p / 24, x = p % 24;
            h2[((b * 16 + l) * 24 + y) * 24 + x] = c[v];
        }
    }
}

// ---------------- maxpool 2x2 + full-vector bitonic sort (2304 -> pad 4096) ----------
__global__ __launch_bounds__(256) void pool_sort_kernel(const float* __restrict__ h2,
                                                        float* __restrict__ sorted) {
    __shared__ float lds[4096];
    const int b = blockIdx.x, tid = threadIdx.x;
    for (int j = tid; j < 2304; j += 256) {
        int c = j / 144, rem = j % 144;
        int y = rem / 12, x = rem % 12;
        const float* p = h2 + ((b * 16 + c) * 24 + 2 * y) * 24 + 2 * x;
        lds[j] = fmaxf(fmaxf(p[0], p[1]), fmaxf(p[24], p[25]));
    }
    for (int j = 2304 + tid; j < 4096; j += 256) lds[j] = SENTF;
    __syncthreads();
    for (int k = 2; k <= 4096; k <<= 1) {
        for (int j = k >> 1; j > 0; j >>= 1) {
            for (int q = tid; q < 2048; q += 256) {
                int e = ((q & ~(j - 1)) << 1) | (q & (j - 1));
                int f = e | j;
                float a = lds[e], c = lds[f];
                bool up = ((e & k) == 0);
                if (up ? (a < c) : (a > c)) { lds[e] = c; lds[f] = a; }
            }
            __syncthreads();
        }
    }
    for (int j = tid; j < 2304; j += 256) sorted[b * 2304 + j] = lds[j];
}

// ---------------- FC1: [512,2304] x [2304,16] via WMMA, one wave per 16-sample tile --
__global__ __launch_bounds__(32) void fc1_kernel(const float* __restrict__ sorted,
                                                 const _Float16* __restrict__ w3pk,
                                                 float* __restrict__ h3) {
    const int tile = blockIdx.x;   // 32 tiles of 16 samples
    const int tid = threadIdx.x;
    const int l = tid & 15;
    const bool hi = tid >= 16;
    const float* arow0 = sorted + (tile * 16 + l) * 2304 + (hi ? 8 : 0);
    const v16h* bvec = (const v16h*)w3pk;
    v8f c = {};
    for (int ch = 0; ch < 72; ++ch) {
        const v4f* ap = (const v4f*)(arow0 + ch * 32);   // 32B-aligned
        v4f f0 = ap[0], f1 = ap[1];                      // K = kb+off .. +7
        v4f f2 = ap[4], f3 = ap[5];                      // K = kb+16+off .. +23
        if (ch + 1 < 72)
            __builtin_prefetch(arow0 + (ch + 1) * 32, 0, 1);  // global_prefetch_b8
        v16h a;
#pragma unroll
        for (int kk = 0; kk < 4; ++kk) {
            a[kk]      = (_Float16)f0[kk];
            a[kk + 4]  = (_Float16)f1[kk];
            a[kk + 8]  = (_Float16)f2[kk];
            a[kk + 12] = (_Float16)f3[kk];
        }
        v16h bm = bvec[ch * 32 + tid];                   // 32B contiguous per lane
        c = __builtin_amdgcn_wmma_f32_16x16x32_f16(false, a, false, bm,
                                                   (short)0, c, false, false);
    }
#pragma unroll
    for (int v = 0; v < 8; ++v) {
        int m = v + (hi ? 8 : 0);
        h3[(tile * 16 + m) * 16 + l] = c[v];
    }
}

// ---------------- FC2: sort-16 + dot-16 per sample ----------------
__global__ __launch_bounds__(256) void fc2_kernel(const float* __restrict__ h3,
                                                  const float* __restrict__ w4,
                                                  float* __restrict__ out) {
    int b = blockIdx.x * 256 + threadIdx.x;
    if (b >= 512) return;
    float v[16];
#pragma unroll
    for (int i = 0; i < 16; ++i) v[i] = h3[b * 16 + i];
#pragma unroll
    for (int i = 0; i < 15; ++i)
#pragma unroll
        for (int j = 0; j < 15 - i; ++j) {
            float a = v[j], c = v[j + 1];
            v[j] = fmaxf(a, c); v[j + 1] = fminf(a, c);
        }
    float acc = 0.f;
#pragma unroll
    for (int i = 0; i < 16; ++i) acc += v[i] * w4[i];
    out[b] = acc;
}

extern "C" void kernel_launch(void* const* d_in, const int* in_sizes, int n_in,
                              void* d_out, int out_size, void* d_ws, size_t ws_size,
                              hipStream_t stream) {
    const float* x  = (const float*)d_in[0];
    const float* w1 = (const float*)d_in[1];
    const float* w2 = (const float*)d_in[2];
    const float* w3 = (const float*)d_in[3];
    const float* w4 = (const float*)d_in[4];
    float* out = (float*)d_out;

    char* ws = (char*)d_ws;
    size_t off = 0;
    auto alloc = [&](size_t bytes) -> void* {
        void* p = ws + off;
        off = (off + bytes + 255) & ~(size_t)255;
        return p;
    };
    float*     h1     = (float*)alloc((size_t)512 * 16 * 26 * 26 * 4);
    float*     h2     = (float*)alloc((size_t)512 * 16 * 24 * 24 * 4);
    float*     sorted = (float*)alloc((size_t)512 * 2304 * 4);
    float*     h3     = (float*)alloc((size_t)512 * 16 * 4);
    _Float16*  w2pk   = (_Float16*)alloc((size_t)5 * 512 * 2);
    _Float16*  w3pk   = (_Float16*)alloc((size_t)72 * 512 * 2);

    initw_kernel<<<(72 * 512 + 255) / 256, 256, 0, stream>>>(w2, w3, w2pk, w3pk);
    l1_kernel<<<(512 * 676 + 255) / 256, 256, 0, stream>>>(x, w1, h1);
    l2_kernel<<<512 * 36, 256, 0, stream>>>(h1, w2pk, h2);
    pool_sort_kernel<<<512, 256, 0, stream>>>(h2, sorted);
    fc1_kernel<<<32, 32, 0, stream>>>(sorted, w3pk, h3);
    fc2_kernel<<<2, 256, 0, stream>>>(h3, w4, out);
}
